// MalConvLowMem_32976758898851
// MI455X (gfx1250) — compile-verified, matrix operand order checked by reading, not run on hardware
//
#include <hip/hip_runtime.h>
#include <hip/hip_bf16.h>

// ---------------------------------------------------------------------------
// MalConv low-mem on MI455X (gfx1250):
//   pass 1: bf16 WMMA gated-conv max scan (argmax per (b,c), packed u64 atomics)
//           64-row M-tiles per workgroup: each B fragment feeds 4 WMMAs,
//           cutting L2 weight traffic 4x (below the 128 MB HBM floor).
//   pass 2: exact fp32 recompute of the 512 winning windows
// ---------------------------------------------------------------------------

typedef __attribute__((ext_vector_type(16))) __bf16 v16bf;
typedef __attribute__((ext_vector_type(8)))  float  v8f;
typedef __attribute__((ext_vector_type(4)))  float  v4f;

#define BB      4
#define TT      1000000
#define EE      8
#define CC      128
#define KSZ     512
#define TO      1953                 // (T - KS)/STRIDE + 1
#define MWIN    (BB * TO)            // 7812 windows
#define KDIM    (EE * KSZ)           // 4096
#define KSTEPS  (KDIM / 32)          // 128 wmma steps
#define SUBT    4                    // 16-row subtiles per workgroup tile
#define MT      (16 * SUBT)          // 64 rows per workgroup
#define MTILES  ((MWIN + MT - 1) / MT)   // 123

// ---------------------------------------------------------------------------
// prep: zero argmax slots, pack [W1|W2] into per-lane WMMA-B fragment layout.
// B frag layout (bf16 16x16x32): lane holds N = lane%16; per-lane j-th value
// is K = kstep*32 + c0 + (j<8 ? j : j+8), c0 = (lane<16 ? 0 : 8).
// Window flat index f -> (e = f%8, k = f/8), so B[f][c] = W[c][f%8][f/8].
// ---------------------------------------------------------------------------
__global__ void __launch_bounds__(256)
malconv_prep(const float* __restrict__ W1, const float* __restrict__ W2,
             unsigned long long* __restrict__ slots, __bf16* __restrict__ bpack) {
    int gid = blockIdx.x * blockDim.x + threadIdx.x;   // 0 .. 65535
    if (gid < BB * CC) slots[gid] = 0ULL;
    int lane  = gid & 31;
    int kstep = (gid >> 5) & (KSTEPS - 1);
    int ns    = gid >> 12;                             // 0..15 N-strip
    const float* W = (ns < 8) ? W1 : W2;
    int c  = ((ns & 7) << 4) + (lane & 15);
    int c0 = (lane < 16) ? 0 : 8;
    __bf16* dst = bpack + (size_t)gid * 16;
#pragma unroll
    for (int j = 0; j < 16; ++j) {
        int kk = kstep * 32 + c0 + (j < 8 ? j : j + 8); // window-flat index f
        int e = kk & 7, k = kk >> 3;
        dst[j] = (__bf16)W[(size_t)c * KDIM + e * KSZ + k];
    }
}

// ---------------------------------------------------------------------------
// scan: one workgroup per 64-window M tile; wave w handles channels
// [16w, 16w+16) for BOTH gates; 4 M-subtiles reuse each B fragment pair.
// ---------------------------------------------------------------------------
__global__ void __launch_bounds__(256)
malconv_scan(const float* __restrict__ x,
             const float* __restrict__ b1, const float* __restrict__ b2,
             const __bf16* __restrict__ bpack,
             unsigned long long* __restrict__ slots) {
    const int tile = blockIdx.x;
    const int lane = threadIdx.x & 31;
    const int w    = threadIdx.x >> 5;          // wave 0..7
    const int c0   = (lane < 16) ? 0 : 8;

    // A-fragment row pointers, one per subtile (A layout: M = lane%16).
    const float* aptr[SUBT];
#pragma unroll
    for (int sub = 0; sub < SUBT; ++sub) {
        int mA = tile * MT + sub * 16 + (lane & 15);
        if (mA > MWIN - 1) mA = MWIN - 1;
        int bA  = mA / TO;
        int toA = mA - bA * TO;
        aptr[sub] = x + ((size_t)bA * TT + (size_t)toA * KSZ) * EE + c0;
    }

    const __bf16* bp1 = bpack + ((size_t)((w    ) * KSTEPS) * 32 + lane) * 16;
    const __bf16* bp2 = bpack + ((size_t)((8 + w) * KSTEPS) * 32 + lane) * 16;

    v8f acc1[SUBT]; v8f acc2[SUBT];
#pragma unroll
    for (int sub = 0; sub < SUBT; ++sub) { acc1[sub] = (v8f){}; acc2[sub] = (v8f){}; }

    for (int s = 0; s < KSTEPS; ++s) {
        v16bf bf1 = *(const v16bf*)(bp1 + (size_t)s * 512);
        v16bf bf2 = *(const v16bf*)(bp2 + (size_t)s * 512);
#pragma unroll
        for (int sub = 0; sub < SUBT; ++sub) {
            const float* ap = aptr[sub] + s * 32;
            v4f a0 = *(const v4f*)(ap);
            v4f a1 = *(const v4f*)(ap + 4);
            v4f a2 = *(const v4f*)(ap + 16);
            v4f a3 = *(const v4f*)(ap + 20);
            __builtin_prefetch(ap + 64, 0, 3);   // global_prefetch_b8: next A chunk

            v16bf af;
            af[0]  = (__bf16)a0[0]; af[1]  = (__bf16)a0[1];
            af[2]  = (__bf16)a0[2]; af[3]  = (__bf16)a0[3];
            af[4]  = (__bf16)a1[0]; af[5]  = (__bf16)a1[1];
            af[6]  = (__bf16)a1[2]; af[7]  = (__bf16)a1[3];
            af[8]  = (__bf16)a2[0]; af[9]  = (__bf16)a2[1];
            af[10] = (__bf16)a2[2]; af[11] = (__bf16)a2[3];
            af[12] = (__bf16)a3[0]; af[13] = (__bf16)a3[1];
            af[14] = (__bf16)a3[2]; af[15] = (__bf16)a3[3];

            acc1[sub] = __builtin_amdgcn_wmma_f32_16x16x32_bf16(
                false, af, false, bf1, (short)0, acc1[sub], false, false);
            acc2[sub] = __builtin_amdgcn_wmma_f32_16x16x32_bf16(
                false, af, false, bf2, (short)0, acc2[sub], false, false);
        }
    }

    // Epilogue: bias + sigmoid gate, then packed (value,to) atomic argmax.
    const int c = (w << 4) + (lane & 15);       // channel (C/D layout: N = lane%16)
    const float b1c = b1[c];
    const float b2c = b2[c];

#pragma unroll
    for (int sub = 0; sub < SUBT; ++sub) {
        int mLo = tile * MT + sub * 16;
        int mHi = mLo + 15; if (mHi > MWIN - 1) mHi = MWIN - 1;
        const bool uniform_b = (mLo / TO) == (mHi / TO);

        float best = -3.402823466e38f;
        int bestM = 0;
#pragma unroll
        for (int r = 0; r < 8; ++r) {
            int m = mLo + r + ((lane < 16) ? 0 : 8);   // C layout: M = r + 8*(lane>=16)
            if (m > MWIN - 1) m = MWIN - 1;
            float g1 = acc1[sub][r] + b1c;
            float g2 = acc2[sub][r] + b2c;
            float g  = g1 * (1.0f / (1.0f + __expf(-g2)));
            if (uniform_b) {
                if (g > best) { best = g; bestM = m; }
            } else {
                int bb = m / TO;
                int to = m - bb * TO;
                unsigned int u = __float_as_uint(g);
                u = (u & 0x80000000u) ? ~u : (u | 0x80000000u);
                unsigned long long pk = ((unsigned long long)u << 32) | (unsigned int)to;
                atomicMax(&slots[bb * CC + c], pk);
            }
        }
        if (uniform_b) {
            int bb = bestM / TO;
            int to = bestM - bb * TO;
            unsigned int u = __float_as_uint(best);
            u = (u & 0x80000000u) ? ~u : (u | 0x80000000u);
            unsigned long long pk = ((unsigned long long)u << 32) | (unsigned int)to;
            atomicMax(&slots[bb * CC + c], pk);
        }
    }
}

// ---------------------------------------------------------------------------
// rescore: one wave per (b,c); exact fp32 dot on the winning window.
// ---------------------------------------------------------------------------
__global__ void __launch_bounds__(256)
malconv_rescore(const float* __restrict__ x,
                const float* __restrict__ W1, const float* __restrict__ b1,
                const float* __restrict__ W2, const float* __restrict__ b2,
                const unsigned long long* __restrict__ slots,
                float* __restrict__ out) {
    int wid  = (blockIdx.x * blockDim.x + threadIdx.x) >> 5;   // 0..511
    int lane = threadIdx.x & 31;
    if (wid >= BB * CC) return;
    int b = wid >> 7, c = wid & (CC - 1);

    int to  = (int)(slots[wid] & 0xFFFFFFFFu);
    long pos = (long)to * KSZ;
    if (pos > (long)(TT - KSZ)) pos = TT - KSZ;                // reference clip

    const float* win = x  + ((size_t)b * TT + pos) * EE;
    const float* w1  = W1 + (size_t)c * KDIM;
    const float* w2  = W2 + (size_t)c * KDIM;

    float a1 = 0.0f, a2 = 0.0f;
    for (int f = lane; f < KDIM; f += 32) {
        float xv = win[f];
        int e = f & 7, k = f >> 3;
        float wi = w1[e * KSZ + k];
        float wj = w2[e * KSZ + k];
        a1 = fmaf(xv, wi, a1);
        a2 = fmaf(xv, wj, a2);
    }
#pragma unroll
    for (int off = 16; off > 0; off >>= 1) {
        a1 += __shfl_xor(a1, off, 32);
        a2 += __shfl_xor(a2, off, 32);
    }
    if (lane == 0) {
        float g1 = a1 + b1[c];
        float g2 = a2 + b2[c];
        out[wid] = g1 * (1.0f / (1.0f + __expf(-g2)));
    }
}

// ---------------------------------------------------------------------------
extern "C" void kernel_launch(void* const* d_in, const int* in_sizes, int n_in,
                              void* d_out, int out_size, void* d_ws, size_t ws_size,
                              hipStream_t stream) {
    const float* x  = (const float*)d_in[0];
    const float* W1 = (const float*)d_in[1];
    const float* b1 = (const float*)d_in[2];
    const float* W2 = (const float*)d_in[3];
    const float* b2 = (const float*)d_in[4];
    float* out = (float*)d_out;

    unsigned long long* slots = (unsigned long long*)d_ws;        // 512 * 8 B
    __bf16* bpack = (__bf16*)((char*)d_ws + 4096);                // 2 MB packed B

    // 1) zero slots + pack weights into WMMA-B fragment layout
    malconv_prep<<<256, 256, 0, stream>>>(W1, W2, slots, bpack);
    // 2) bf16 WMMA gated max scan over all 7812 windows (64-row tiles)
    malconv_scan<<<MTILES, 256, 0, stream>>>(x, b1, b2, bpack, slots);
    // 3) exact fp32 recompute of 512 winners
    malconv_rescore<<<64, 256, 0, stream>>>(x, W1, b1, W2, b2, slots, out);
}
